// TestModel_21809843929939
// MI455X (gfx1250) — compile-verified, matrix operand order checked by reading, not run on hardware
//
#include <hip/hip_runtime.h>
#include <hip/hip_bf16.h>

// ---------------------------------------------------------------------------
// Deformable conv v2 forward for MI455X (gfx1250, wave32, WMMA).
// x:     (4, 64, 128, 128) fp32
// w_off: (18, 64, 3, 3)    fp32   -> offset conv (fp32 VALU)
// w_dc:  (128, 64, 3, 3)   fp32   -> main conv as bf16 WMMA GEMM
//                                    M=128(o) x N=65536(b*h*w) x K=576(c*9)
// out:   (4, 128, 128, 128) fp32
// ---------------------------------------------------------------------------

#define BATCH 4
#define CIN   64
#define HH    128
#define WW    128
#define OC    128
#define NK    9          // 3x3 taps
#define KTOT  (CIN * NK) // 576 = GEMM K
#define PIXB  (HH * WW)  // 16384 pixels per batch
#define PTILE 32         // pixels per block (two 16-wide WMMA N tiles)

typedef __attribute__((ext_vector_type(16))) __bf16 v16bf;
typedef __attribute__((ext_vector_type(8)))  __bf16 v8bf;
typedef __attribute__((ext_vector_type(8)))  float  v8f;

union FragU {
    struct { v8bf lo, hi; } p;
    v16bf v;
};

__device__ __forceinline__ unsigned short f2bf(float f) {
    unsigned u = __builtin_bit_cast(unsigned, f);
    u += 0x7FFFu + ((u >> 16) & 1u);          // round-to-nearest-even
    return (unsigned short)(u >> 16);
}

// ---------------------------------------------------------------------------
// Kernel 1: convert w_dc (fp32, [o][c][3][3] == [o][k], k = c*9+n) to bf16.
// ---------------------------------------------------------------------------
__global__ void k_convert_w(const float* __restrict__ w_dc,
                            unsigned short* __restrict__ w_bf) {
    int i = blockIdx.x * blockDim.x + threadIdx.x;
    if (i < OC * KTOT) w_bf[i] = f2bf(w_dc[i]);
}

// ---------------------------------------------------------------------------
// Kernel 2: offset conv. 3x3, pad 1, 18 output channels, fp32 VALU.
// One block = 16x16 pixel tile; w_off staged in LDS.
// Output layout: off_buf[(b*PIXB + h*WW + w)*18 + m]  (pixel-major).
// ---------------------------------------------------------------------------
__global__ void k_offset_conv(const float* __restrict__ x,
                              const float* __restrict__ w_off,
                              float* __restrict__ off_buf) {
    __shared__ float sW[18 * CIN * NK];  // 10368 floats = 41.5 KB
    int tid = threadIdx.x;
    for (int i = tid; i < 18 * CIN * NK; i += 256) sW[i] = w_off[i];
    __syncthreads();

    int b  = blockIdx.x >> 6;          // 64 tiles per batch
    int t  = blockIdx.x & 63;
    int h0 = (t >> 3) << 4;
    int w0 = (t & 7) << 4;
    int hh = h0 + (tid >> 4);
    int ww = w0 + (tid & 15);

    float acc[18];
#pragma unroll
    for (int m = 0; m < 18; ++m) acc[m] = 0.0f;

    for (int c = 0; c < CIN; ++c) {
        const float* xc = x + (((size_t)b * CIN + c) << 14);
        float v[NK];
#pragma unroll
        for (int dy = -1; dy <= 1; ++dy) {
#pragma unroll
            for (int dx = -1; dx <= 1; ++dx) {
                int yy = hh + dy, xx = ww + dx;
                bool ok = (yy >= 0) & (yy < HH) & (xx >= 0) & (xx < WW);
                v[(dy + 1) * 3 + (dx + 1)] = ok ? xc[yy * WW + xx] : 0.0f;
            }
        }
#pragma unroll
        for (int m = 0; m < 18; ++m) {
            const float* wr = &sW[(m * CIN + c) * NK];
#pragma unroll
            for (int n = 0; n < NK; ++n) acc[m] = fmaf(wr[n], v[n], acc[m]);
        }
    }

    float* op = off_buf + ((size_t)(b * PIXB + hh * WW + ww)) * 18;
#pragma unroll
    for (int m = 0; m < 18; ++m) op[m] = acc[m];
}

// ---------------------------------------------------------------------------
// Kernel 3: bilinear gather + WMMA GEMM.
// Block = 256 threads = 8 wave32. One block owns a strip of 32 pixels
// (same b, same h, 32 consecutive w) and ALL 128 output channels.
//   Stage 1: 288 (pixel, tap) items -> 4 corner indices + 4 bilinear
//            weights each, into LDS.
//   Stage 2: gather 32x576 bf16 B-tile into LDS. Each work unit fixes one
//            (pixel, tap) and sweeps 16 channels, so the 8-word param fetch
//            is amortized over 64 scattered x-loads (params stay in VGPRs).
//   Stage 3: wave i owns o-tile i; per K-step it loads its A fragment ONCE
//            and feeds BOTH 16-pixel B tiles -> 36 WMMAs per wave.
// ---------------------------------------------------------------------------
__global__ void k_deform_wmma(const float* __restrict__ x,
                              const unsigned short* __restrict__ w_bf,
                              const float* __restrict__ off_buf,
                              float* __restrict__ out) {
    __shared__ __align__(16) unsigned short sXoff[PTILE * KTOT]; // 36 KB bf16 B tiles
    __shared__ int   sIdx[PTILE * NK * 4];                       // corner flat idx (-1 = zero)
    __shared__ float sWgt[PTILE * NK * 4];                       // bilinear weights

    const int tid = threadIdx.x;
    const int b   = blockIdx.x >> 9;           // 512 strips per batch
    const int t   = blockIdx.x & 511;
    const int hh  = t >> 2;                    // row 0..127
    const int w0  = (t & 3) << 5;              // col base 0,32,64,96
    const int pixbase = hh * WW + w0;          // in-batch pixel index of strip start

    // ---- Stage 1: bilinear parameters per (pixel, tap) --------------------
    for (int item = tid; item < PTILE * NK; item += 256) {
        int pix = item / NK;
        int n   = item - pix * NK;
        int rr  = n / 3 - 1;
        int cc  = n % 3 - 1;
        const float* ob = off_buf + ((size_t)(b * PIXB + pixbase + pix)) * 18;
        float offy = ob[2 * n];         // even channels = row offsets (after perm)
        float offx = ob[2 * n + 1];

        // padded coordinates (hp = wp = 130), pixel centers at 1..128
        float py = (float)(hh + 1 + rr) + offy;
        float px = (float)(w0 + pix + 1 + cc) + offx;

        // reference edge handling: snap-to-floor outside valid band, clamp
        if (py < 1.0f || py > 128.0f) py = floorf(py);
        if (px < 1.0f || px > 128.0f) px = floorf(px);
        py = fminf(fmaxf(py, 0.0f), 129.0f);
        px = fminf(fmaxf(px, 0.0f), 129.0f);

        float fy = floorf(py), fx = floorf(px);
        int qy0 = (int)fy, qx0 = (int)fx;
        int qy1 = min(qy0 + 1, 129), qx1 = min(qx0 + 1, 129);
        float ly = py - fy, lx = px - fx;

        int   qy[4] = { qy0, qy0, qy1, qy1 };
        int   qx[4] = { qx0, qx1, qx0, qx1 };
        float gw[4] = { (1.0f - ly) * (1.0f - lx), (1.0f - ly) * lx,
                        ly * (1.0f - lx),          ly * lx };
        int base = item * 4;
#pragma unroll
        for (int q = 0; q < 4; ++q) {
            // padded border (ring of zeros) -> idx = -1
            bool inb = (qy[q] >= 1) & (qy[q] <= HH) & (qx[q] >= 1) & (qx[q] <= WW);
            sIdx[base + q] = inb ? ((qy[q] - 1) * WW + (qx[q] - 1)) : -1;
            sWgt[base + q] = gw[q];
        }
    }
    __syncthreads();

    // ---- Stage 2: gather PTILE x 576 bf16 tile into LDS -------------------
    // unit u: item = u>>2 -> (pix, n); cg = u&3 -> channel group of 16.
    // B element layout: sXoff[pix*KTOT + c*9 + n]  (k = c*9+n matches w_dc).
    for (int u = tid; u < PTILE * NK * 4; u += 256) {
        int item = u >> 2;
        int cg   = u & 3;
        int pix  = item / NK;
        int n    = item - pix * NK;

        int   id[4];
        float gw[4];
        int pb = item * 4;
#pragma unroll
        for (int q = 0; q < 4; ++q) { id[q] = sIdx[pb + q]; gw[q] = sWgt[pb + q]; }

        int c0 = cg << 4;
        const float* xc = x + (((size_t)b * CIN + c0) << 14);
        unsigned short* dst = sXoff + pix * KTOT + c0 * NK + n;
#pragma unroll 4
        for (int ci = 0; ci < 16; ++ci) {
            float acc = 0.0f;
#pragma unroll
            for (int q = 0; q < 4; ++q) {
                float xv = (id[q] >= 0) ? xc[id[q]] : 0.0f;
                acc = fmaf(xv, gw[q], acc);
            }
            dst[ci * NK] = f2bf(acc);
            xc += PIXB;                      // next channel plane
        }
    }
    __syncthreads();

    // ---- Stage 3: WMMA GEMM, one 16-row o-tile per wave, 2 N tiles --------
    const int wave    = tid >> 5;      // 0..7 -> o-tile
    const int lane    = tid & 31;
    const int row     = lane & 15;     // A: o-row within tile / B: pixel column
    const int halfsel = lane >> 4;     // selects K sub-chunk per 16-bit layout

    const unsigned short* wtile = w_bf + (size_t)(wave * 16 + row) * KTOT;
    __builtin_prefetch(wtile, 0, 3);   // global_prefetch_b8 on the weight stream

    const unsigned short* brow0 = sXoff + row * KTOT;            // pixels 0..15
    const unsigned short* brow1 = sXoff + (16 + row) * KTOT;     // pixels 16..31

    v8f acc0 = {}, acc1 = {};
#pragma unroll
    for (int ks = 0; ks < KTOT / 32; ++ks) {
        int k0 = ks * 32 + halfsel * 8;
        FragU a, b0, b1;
        a.p.lo  = *(const v8bf*)(wtile + k0);
        a.p.hi  = *(const v8bf*)(wtile + k0 + 16);
        b0.p.lo = *(const v8bf*)(brow0 + k0);
        b0.p.hi = *(const v8bf*)(brow0 + k0 + 16);
        b1.p.lo = *(const v8bf*)(brow1 + k0);
        b1.p.hi = *(const v8bf*)(brow1 + k0 + 16);
        acc0 = __builtin_amdgcn_wmma_f32_16x16x32_bf16(
            false, a.v, false, b0.v, (short)0, acc0, false, false);
        acc1 = __builtin_amdgcn_wmma_f32_16x16x32_bf16(
            false, a.v, false, b1.v, (short)0, acc1, false, false);
    }

    // C/D layout: VGPR r -> M = halfsel*8 + r, lane&15 -> N (pixel)
    int o0 = wave * 16 + halfsel * 8;
#pragma unroll
    for (int r = 0; r < 8; ++r) {
        size_t orow = ((size_t)(b * OC + o0 + r)) << 14;
        out[orow + pixbase + row]      = acc0[r];
        out[orow + pixbase + 16 + row] = acc1[r];
    }
}

// ---------------------------------------------------------------------------
// Launch
// ---------------------------------------------------------------------------
extern "C" void kernel_launch(void* const* d_in, const int* in_sizes, int n_in,
                              void* d_out, int out_size, void* d_ws, size_t ws_size,
                              hipStream_t stream) {
    (void)in_sizes; (void)n_in; (void)out_size; (void)ws_size;
    const float* x     = (const float*)d_in[0];
    const float* w_off = (const float*)d_in[1];
    const float* w_dc  = (const float*)d_in[2];
    float* out = (float*)d_out;

    // workspace layout: [0, 256KB) bf16 weights (144KB), then offset buffer (4.5MB)
    unsigned short* w_bf = (unsigned short*)d_ws;
    float* off_buf = (float*)((char*)d_ws + 256 * 1024);

    k_convert_w<<<(OC * KTOT + 255) / 256, 256, 0, stream>>>(w_dc, w_bf);
    k_offset_conv<<<BATCH * 64, 256, 0, stream>>>(x, w_off, off_buf);
    k_deform_wmma<<<BATCH * 512, 256, 0, stream>>>(x, w_bf, off_buf, out);
}
